// ContextualAttention_51539608052
// MI455X (gfx1250) — compile-verified, matrix operand order checked by reading, not run on hardware
//
#include <hip/hip_runtime.h>

typedef _Float16 half_t;
typedef __attribute__((ext_vector_type(8)))  _Float16 v8h;
typedef __attribute__((ext_vector_type(16))) _Float16 v16h;
typedef __attribute__((ext_vector_type(8)))  float    v8f;
typedef __attribute__((ext_vector_type(4)))  unsigned int u32x4;
typedef __attribute__((ext_vector_type(4)))  int          i32x4;
typedef __attribute__((ext_vector_type(8)))  int          i32x8;

#define HAVE_TDM __has_builtin(__builtin_amdgcn_tensor_load_to_lds)

namespace {
constexpr int CH = 64;     // channels (inner dim of score GEMM, M dim of weight GEMM)
constexpr int NQ = 4096;   // H*W  (keys K and queries Q)
constexpr int BQ = 128;    // q columns per workgroup (8 waves x 16)
constexpr int KT = 64;     // k-tile staged in LDS per iteration
constexpr float NEG_INF = -3.0e38f;

union Frag16 {
  v16h   v;
  v8h    h[2];
  half_t e[16];
};
} // namespace

#if HAVE_TDM
// Issue a TDM DMA of a 2D tile: 64 (contiguous k) x 64 (c rows, stride 4096 floats)
// of f32 from global memory into LDS at byte offset lds_off. Tracked by TENSORcnt.
__device__ __forceinline__ void tdm_load_tile(unsigned lds_off, const float* gsrc) {
  unsigned long long ga = (unsigned long long)(size_t)gsrc;
  u32x4 g0;
  g0.x = 1u;                                               // count=1 (valid user D#)
  g0.y = lds_off;                                          // lds_addr (bytes)
  g0.z = (unsigned)(ga & 0xffffffffu);                     // global_addr[31:0]
  g0.w = (unsigned)((ga >> 32) & 0x01ffffffu) | (2u << 30);// global_addr[56:32] | type=2
  i32x8 g1;
  g1[0] = (int)(2u << 16);        // workgroup_mask=0, data_size=2 (4 bytes), no pad/iter
  g1[1] = (int)(4096u << 16);     // tensor_dim0[15:0]=4096 in bits[63:48]
  g1[2] = (int)(64u << 16);       // tensor_dim0 hi=0; tensor_dim1[15:0]=64 in bits[95:80]
  g1[3] = (int)(64u << 16);       // tensor_dim1 hi=0; tile_dim0=64 in bits[127:112]
  g1[4] = (int)(64u);             // tile_dim1=64; tile_dim2=0
  g1[5] = (int)(4096u);           // tensor_dim0_stride[31:0] = 4096
  g1[6] = 0;                      // stride hi / tensor_dim1_stride lo (unused, 2D)
  g1[7] = 0;
  i32x4 g2 = {0, 0, 0, 0};
  i32x4 g3 = {0, 0, 0, 0};
#if __clang_major__ >= 23
  i32x8 g4 = {0, 0, 0, 0, 0, 0, 0, 0};
  __builtin_amdgcn_tensor_load_to_lds(g0, g1, g2, g3, g4, 0);
#else
  __builtin_amdgcn_tensor_load_to_lds(g0, g1, g2, g3, 0);
#endif
}
#endif // HAVE_TDM

__global__ __launch_bounds__(256, 1)
void ctx_attn_flash_wmma(const float* __restrict__ bg,
                         const float* __restrict__ fg,
                         const float* __restrict__ mask,
                         float* __restrict__ out) {
#if HAVE_TDM
  __shared__ alignas(16) float  rawt[2][CH][KT];  // double-buffered raw bg tiles (TDM dest)
#endif
  __shared__ alignas(16) half_t nbg[KT][CH];      // normalized bg tile, [k][c]
  __shared__ alignas(16) half_t bgh[CH][KT];      // raw bg tile (f16), [c][k]
  __shared__ float ksum[KT];

  const int b    = blockIdx.y;
  const int tid  = (int)threadIdx.x;
  const int lane = tid & 31;
  const int wave = tid >> 5;
  const int lq   = lane & 15;   // q column within the wave tile
  const int hid  = lane >> 4;   // 0: lanes 0-15, 1: lanes 16-31
  const int q    = blockIdx.x * BQ + wave * 16 + lq;

  const float* bgB = bg + (size_t)b * CH * NQ;
  const float* fgB = fg + (size_t)b * CH * NQ;

#if HAVE_TDM
  const unsigned raw_lds0 = (unsigned)(size_t)&rawt[0][0][0];
  const unsigned raw_lds1 = (unsigned)(size_t)&rawt[1][0][0];
  // Kick off the DMA for tile 0 before the prologue; it overlaps the fg work below.
  if (wave == 0) tdm_load_tile(raw_lds0, bgB);
#endif

  // ---------- Prologue: normalized-fg B fragments (K_inner = channel) ----------
  // B layout (16-bit, 32x16): lanes 0-15 hold K(c)=0..7 in e[0..7], 16..23 in e[8..15];
  // lanes 16-31 hold c=8..15 and 24..31. Fragment 1 adds +32 to c.
  Frag16 fB[2];
  float  fv[2][16];
  float  fsq = 0.f;
  #pragma unroll
  for (int f = 0; f < 2; ++f) {
    #pragma unroll
    for (int p = 0; p < 2; ++p) {
      #pragma unroll
      for (int j = 0; j < 8; ++j) {
        int   c = 32 * f + 16 * p + 8 * hid + j;
        float v = fgB[(size_t)c * NQ + q];
        fv[f][8 * p + j] = v;
        fsq += v * v;
      }
    }
  }
  fsq += __shfl_xor(fsq, 16, 32);                 // lane pair covers all 64 channels
  float finv = 1.0f / fmaxf(sqrtf(fsq), 1e-12f);  // matches F.normalize eps semantics
  #pragma unroll
  for (int f = 0; f < 2; ++f)
    #pragma unroll
    for (int i = 0; i < 16; ++i)
      fB[f].e[i] = (half_t)(fv[f][i] * finv);

  // ---------- Online-softmax state + output accumulators ----------
  v8f acc[4];                                     // 64(c) x 16(q) fragment, 4 c-chunks
  #pragma unroll
  for (int cc = 0; cc < 4; ++cc) { v8f z = {}; acc[cc] = z; }
  float m_run = NEG_INF;
  float l_run = 0.f;

  const int lk = tid & 63;          // this thread's key column in the staged tile
  const int cb = (tid >> 6) * 16;   // this thread's channel sub-block
  constexpr int NT = NQ / KT;       // 64 key tiles

  // ---------- Flash loop over keys ----------
  for (int t = 0; t < NT; ++t) {
    const int kt = t * KT;
    __syncthreads();   // (A) prior-iter LDS consumers done (nbg/bgh and next raw buffer)

#if HAVE_TDM
    if (wave == 0) {
      if (t + 1 < NT) {             // prefetch next tile while this one is consumed
        tdm_load_tile(((t + 1) & 1) ? raw_lds1 : raw_lds0, bgB + (kt + KT));
        __builtin_amdgcn_s_wait_tensorcnt(1);   // oldest (current tile) DMA done
      } else {
        __builtin_amdgcn_s_wait_tensorcnt(0);   // drain
      }
    }
#endif
    if (tid < KT) ksum[tid] = 0.f;
    __syncthreads();   // (B) raw tile visible to all waves, ksum zeroed

    float v[16];
    float ps = 0.f;
#if HAVE_TDM
    const float* rw = &rawt[t & 1][0][0];
    #pragma unroll
    for (int i = 0; i < 16; ++i) {
      v[i] = rw[(cb + i) * KT + lk];
      ps += v[i] * v[i];
    }
#else
    #pragma unroll
    for (int i = 0; i < 16; ++i) {  // coalesced: consecutive tids -> consecutive k
      v[i] = bgB[(size_t)(cb + i) * NQ + kt + lk];
      ps += v[i] * v[i];
    }
#endif
    atomicAdd(&ksum[lk], ps);       // per-key ||.||^2 across 4 channel sub-blocks
    __syncthreads();   // (C)
    float kinv = 1.0f / fmaxf(sqrtf(ksum[lk]), 1e-12f);
    #pragma unroll
    for (int i = 0; i < 16; ++i) {
      nbg[lk][cb + i] = (half_t)(v[i] * kinv);
      bgh[cb + i][lk] = (half_t)v[i];
    }
    __syncthreads();   // (D)

    #pragma unroll
    for (int s = 0; s < 2; ++s) {   // two 32-key steps per tile
      const int kb = s * 32;

      // --- scores S[k,q] = nbg^T(c,k) * nfg(c,q): two 16-row halves ---
      v8f S[2];
      #pragma unroll
      for (int h = 0; h < 2; ++h) {
        const int krow = kb + 16 * h + lq;   // A layout: M row = lane%16
        Frag16 A0, A1;
        A0.h[0] = *(const v8h*)&nbg[krow][ 0 + 8 * hid];
        A0.h[1] = *(const v8h*)&nbg[krow][16 + 8 * hid];
        A1.h[0] = *(const v8h*)&nbg[krow][32 + 8 * hid];
        A1.h[1] = *(const v8h*)&nbg[krow][48 + 8 * hid];
        v8f z = {};
        v8f d = __builtin_amdgcn_wmma_f32_16x16x32_f16(
                    false, A0.v, false, fB[0].v, (short)0, z, false, false);
        d     = __builtin_amdgcn_wmma_f32_16x16x32_f16(
                    false, A1.v, false, fB[1].v, (short)0, d, false, false);
        S[h] = d;
      }

      // --- online softmax over this 32-key slab (per q column) ---
      float tmax = NEG_INF;
      #pragma unroll
      for (int h = 0; h < 2; ++h)
        #pragma unroll
        for (int r = 0; r < 8; ++r) tmax = fmaxf(tmax, S[h][r]);
      tmax = fmaxf(tmax, __shfl_xor(tmax, 16, 32));
      const float m_new = fmaxf(m_run, tmax);
      const float scale = __expf(m_run - m_new);
      m_run = m_new;
      #pragma unroll
      for (int cc = 0; cc < 4; ++cc) acc[cc] = acc[cc] * scale;

      // exp + pack: score D-fragment (lane=q, rows 8*hid+r) is *exactly* the
      // f16 B-fragment layout of the next GEMM -> per-lane pack, no shuffles.
      Frag16 P;
      float  psum = 0.f;
      #pragma unroll
      for (int h = 0; h < 2; ++h)
        #pragma unroll
        for (int r = 0; r < 8; ++r) {
          float p = __expf(S[h][r] - m_new);
          psum += p;
          P.e[8 * h + r] = (half_t)p;
        }
      psum += __shfl_xor(psum, 16, 32);
      l_run = l_run * scale + psum;

      // --- attended(c,q) += bg(c,k) * P(k,q), inner k = 32 ---
      #pragma unroll
      for (int cc = 0; cc < 4; ++cc) {
        const int crow = 16 * cc + lq;
        Frag16 A;
        A.h[0] = *(const v8h*)&bgh[crow][kb +  0 + 8 * hid];
        A.h[1] = *(const v8h*)&bgh[crow][kb + 16 + 8 * hid];
        acc[cc] = __builtin_amdgcn_wmma_f32_16x16x32_f16(
                      false, A.v, false, P.v, (short)0, acc[cc], false, false);
      }
    }
  }

  // ---------- Epilogue: normalize softmax, mask blend, store ----------
  const float linv = 1.0f / l_run;
  const float mk   = mask[(size_t)b * NQ + q];
  float* outB = out + (size_t)b * CH * NQ;
  #pragma unroll
  for (int cc = 0; cc < 4; ++cc)
    #pragma unroll
    for (int r = 0; r < 8; ++r) {
      const int    c   = 16 * cc + 8 * hid + r;
      const size_t idx = (size_t)c * NQ + q;
      const float  att = acc[cc][r] * linv;
      outB[idx] = fgB[idx] * (1.0f - mk) + att * mk;
    }
}

extern "C" void kernel_launch(void* const* d_in, const int* in_sizes, int n_in,
                              void* d_out, int out_size, void* d_ws, size_t ws_size,
                              hipStream_t stream) {
  const float* bg   = (const float*)d_in[0];
  const float* fg   = (const float*)d_in[1];
  const float* mask = (const float*)d_in[2];
  float*       out  = (float*)d_out;
  const int B = in_sizes[2] / NQ;          // mask is [B,1,H,W]
  dim3 grid(NQ / BQ, B);
  ctx_attn_flash_wmma<<<grid, 256, 0, stream>>>(bg, fg, mask, out);
}
